// SA_affinity_attention_72653666779470
// MI455X (gfx1250) — compile-verified, matrix operand order checked by reading, not run on hardware
//
#include <hip/hip_runtime.h>
#include <hip/hip_bf16.h>

typedef __attribute__((ext_vector_type(16))) __bf16 bf16x16;
typedef __attribute__((ext_vector_type(8)))  __bf16 bf16x8;
typedef __attribute__((ext_vector_type(8)))  float  f32x8;

#define ERP_DIM 512
#define N_TOK   4096
#define BATCH   4
#define ATTN_SCALE 0.04419417382415922f   // 512^-0.5

// CDNA5 16-bit 16x32 A/B fragment (ISA 7.12.2): lane = row (mod 16); per lane
// the 16 bf16 elements are two contiguous 8-element runs at k = h*8 and
// k = h*8 + 16 (h = lane>>4). With bf16 source data each run is one 16-byte
// load => a fragment is exactly two b128 loads, zero VALU.
__device__ __forceinline__ bf16x16 load_frag_bf16(const __bf16* base, int ld,
                                                  int lane) {
  const __bf16* rowp =
      base + (size_t)(lane & 15) * ld + (((lane >> 4) & 1) << 3);
  const bf16x8 lo = *reinterpret_cast<const bf16x8*>(rowp);
  const bf16x8 hi = *reinterpret_cast<const bf16x8*>(rowp + 16);
  return __builtin_shufflevector(lo, hi, 0, 1, 2, 3, 4, 5, 6, 7, 8, 9, 10, 11,
                                 12, 13, 14, 15);
}

// ---------------------------------------------------------------------------
// One-shot f32 -> bf16 conversion (inputs + weights), 8 elems/thread.
// ---------------------------------------------------------------------------
__global__ void f32_to_bf16_kernel(const float* __restrict__ src,
                                   __bf16* __restrict__ dst, int n) {
  const int i = (blockIdx.x * blockDim.x + threadIdx.x) * 8;
  if (i >= n) return;
  const float4 a = *reinterpret_cast<const float4*>(src + i);
  const float4 b = *reinterpret_cast<const float4*>(src + i + 4);
  bf16x8 o;
  o[0] = (__bf16)a.x; o[1] = (__bf16)a.y; o[2] = (__bf16)a.z; o[3] = (__bf16)a.w;
  o[4] = (__bf16)b.x; o[5] = (__bf16)b.y; o[6] = (__bf16)b.z; o[7] = (__bf16)b.w;
  *reinterpret_cast<bf16x8*>(dst + i) = o;
}

// ---------------------------------------------------------------------------
// Projection: Y[m, d] = sum_c X[m, c] * W[d, c] (+ bias[d]); X, W bf16.
// Block = 128 threads (4 waves); each wave owns a 16x64 output tile.
// ---------------------------------------------------------------------------
template <bool OUT_BF16>
__global__ void gemm_bf16_wmma_kernel(const __bf16* __restrict__ X,
                                      const __bf16* __restrict__ W,
                                      const float* __restrict__ bias,
                                      void* __restrict__ Yv, int C, int D) {
  const int lane = threadIdx.x & 31;
  const int wave = threadIdx.x >> 5;
  const int m0 = blockIdx.y * 64 + wave * 16;
  const int d0 = blockIdx.x * 64;

  f32x8 acc[4] = {};
  const __bf16* xrow = X + (size_t)m0 * C;
  for (int kc = 0; kc < C; kc += 32) {
    __builtin_prefetch(xrow + (size_t)(lane & 15) * C + kc + 64, 0, 0);
    const bf16x16 a = load_frag_bf16(xrow + kc, C, lane);
#pragma unroll
    for (int t = 0; t < 4; ++t) {
      const bf16x16 b =
          load_frag_bf16(W + (size_t)(d0 + 16 * t) * C + kc, C, lane);
      acc[t] = __builtin_amdgcn_wmma_f32_16x16x32_bf16(
          false, a, false, b, (short)0, acc[t], false, false);
    }
  }

  const int n  = lane & 15;
  const int mB = (lane >> 4) ? 8 : 0;
#pragma unroll
  for (int t = 0; t < 4; ++t) {
    const int col = d0 + 16 * t + n;
    const float bv = bias ? bias[col] : 0.0f;
#pragma unroll
    for (int r = 0; r < 8; ++r) {
      const size_t idx = (size_t)(m0 + mB + r) * D + col;
      if constexpr (OUT_BF16)
        ((__bf16*)Yv)[idx] = (__bf16)(acc[t][r] + bv);
      else
        ((float*)Yv)[idx] = acc[t][r] + bv;
    }
  }
}

// ---------------------------------------------------------------------------
// KV projection, d' in [0,1024):  d' < 512 -> K[g, d'] (row-major, bf16)
//                                 d' >= 512 -> VT[b, d'-512, m] (bf16)
// ---------------------------------------------------------------------------
__global__ void gemm_kv_wmma_kernel(const __bf16* __restrict__ KV,
                                    const __bf16* __restrict__ W,
                                    __bf16* __restrict__ Kout,
                                    __bf16* __restrict__ VTout) {
  const int C = ERP_DIM;
  const int lane = threadIdx.x & 31;
  const int wave = threadIdx.x >> 5;
  const int m0 = blockIdx.y * 64 + wave * 16;
  const int d0 = blockIdx.x * 64;

  f32x8 acc[4] = {};
  const __bf16* xrow = KV + (size_t)m0 * C;
  for (int kc = 0; kc < C; kc += 32) {
    __builtin_prefetch(xrow + (size_t)(lane & 15) * C + kc + 64, 0, 0);
    const bf16x16 a = load_frag_bf16(xrow + kc, C, lane);
#pragma unroll
    for (int t = 0; t < 4; ++t) {
      const bf16x16 b =
          load_frag_bf16(W + (size_t)(d0 + 16 * t) * C + kc, C, lane);
      acc[t] = __builtin_amdgcn_wmma_f32_16x16x32_bf16(
          false, a, false, b, (short)0, acc[t], false, false);
    }
  }

  const int n  = lane & 15;
  const int mB = (lane >> 4) ? 8 : 0;
#pragma unroll
  for (int t = 0; t < 4; ++t) {
    const int dd = d0 + 16 * t + n;
#pragma unroll
    for (int r = 0; r < 8; ++r) {
      const int g = m0 + mB + r;          // global kv row = b*4096 + m
      const int b = g >> 12, m = g & 4095;
      const __bf16 v = (__bf16)acc[t][r];
      if (dd < ERP_DIM)
        Kout[(size_t)g * ERP_DIM + dd] = v;
      else
        VTout[((size_t)b * ERP_DIM + (dd - ERP_DIM)) * N_TOK + m] = v;
    }
  }
}

// ---------------------------------------------------------------------------
// Attention: block = (batch b, 16 query rows); 256 threads = 8 waves.
// 256 KB dynamic LDS holds the 16x4096 f32 score block (CDNA5: 320 KB/WGP).
//   Phase 1: S = (Q K^T) * scale -> LDS   (Q bf16 frags resident in VGPRs)
//   Phase 2: f32 row softmax; normalize pass compacts P to bf16 IN PLACE
//            (write byte 2c < read byte 4c, wave-lockstep => hazard-free)
//   Phase 3: O = P * V via bf16 ds/global b128 frags; packed transposed store
//            implements the reference's swapaxes(1,2).reshape for free.
// ---------------------------------------------------------------------------
__global__ void attn_lds_wmma_kernel(const __bf16* __restrict__ QP,
                                     const __bf16* __restrict__ Kmat,
                                     const __bf16* __restrict__ VT,
                                     __bf16* __restrict__ ONT) {
  extern __shared__ float lds_s[];                  // 16 * 4096 f32
  const int lane  = threadIdx.x & 31;
  const int wave  = threadIdx.x >> 5;               // 0..7
  const int b     = blockIdx.x >> 8;
  const int n0    = (blockIdx.x & 255) * 16;
  const int ncol  = lane & 15;
  const int mB    = (lane >> 4) ? 8 : 0;

  // Resident Q fragments: 16 chunks of K=32 covering D=512 (128 VGPRs).
  bf16x16 qf[16];
  {
    const __bf16* qbase = QP + ((size_t)b * N_TOK + n0) * ERP_DIM;
#pragma unroll
    for (int ch = 0; ch < 16; ++ch)
      qf[ch] = load_frag_bf16(qbase + ch * 32, ERP_DIM, lane);
  }

  // ---- Phase 1: scores into LDS --------------------------------------
  for (int mt = wave; mt < N_TOK / 16; mt += 8) {
    const int m0 = mt * 16;
    const __bf16* kbase = Kmat + ((size_t)b * N_TOK + m0) * ERP_DIM;
    f32x8 acc = {};
#pragma unroll
    for (int ch = 0; ch < 16; ++ch) {
      const bf16x16 kf = load_frag_bf16(kbase + ch * 32, ERP_DIM, lane);
      acc = __builtin_amdgcn_wmma_f32_16x16x32_bf16(
          false, qf[ch], false, kf, (short)0, acc, false, false);
    }
#pragma unroll
    for (int r = 0; r < 8; ++r)
      lds_s[(size_t)(mB + r) * N_TOK + m0 + ncol] = acc[r] * ATTN_SCALE;
  }
  __syncthreads();

  // ---- Phase 2: softmax + in-place bf16 compaction (wave owns 2 rows) -
  for (int rr = 0; rr < 2; ++rr) {
    const int row = wave * 2 + rr;
    float* rowp = lds_s + (size_t)row * N_TOK;
    float mx = -3.4e38f;
    for (int c = lane; c < N_TOK; c += 32) mx = fmaxf(mx, rowp[c]);
#pragma unroll
    for (int off = 16; off > 0; off >>= 1) mx = fmaxf(mx, __shfl_xor(mx, off, 32));
    float sum = 0.f;
    for (int c = lane; c < N_TOK; c += 32) {
      const float e = __expf(rowp[c] - mx);
      rowp[c] = e;
      sum += e;
    }
#pragma unroll
    for (int off = 16; off > 0; off >>= 1) sum += __shfl_xor(sum, off, 32);
    const float inv = 1.0f / sum;
    __bf16* rowb = reinterpret_cast<__bf16*>(rowp);
    for (int c = lane; c < N_TOK; c += 32) {
      const __bf16 pb = (__bf16)(rowp[c] * inv);
      __builtin_memcpy((void*)(rowb + c), &pb, sizeof(__bf16));  // alias-safe
    }
  }
  __syncthreads();

  // ---- Phase 3: O = P * V; wave owns 64 output channels ---------------
  // Compacted P: row r at byte offset r*16384 => ld = 8192 bf16 elems.
  const __bf16* plds = reinterpret_cast<const __bf16*>(lds_s);
  const int c0 = wave * 64;
  f32x8 acc[4] = {};
  for (int mc = 0; mc < N_TOK / 32; ++mc) {
    const bf16x16 pf = load_frag_bf16(plds + mc * 32, 2 * N_TOK, lane);
#pragma unroll
    for (int t = 0; t < 4; ++t) {
      const __bf16* vbase =
          VT + ((size_t)b * ERP_DIM + c0 + 16 * t) * N_TOK + mc * 32;
      const bf16x16 vf = load_frag_bf16(vbase, N_TOK, lane);
      acc[t] = __builtin_amdgcn_wmma_f32_16x16x32_bf16(
          false, pf, false, vf, (short)0, acc[t], false, false);
    }
  }
  // Packed transposed store: per lane, 8 consecutive n positions -> 1 b128.
#pragma unroll
  for (int t = 0; t < 4; ++t) {
    const int d = c0 + 16 * t + ncol;
    bf16x8 o;
#pragma unroll
    for (int r = 0; r < 8; ++r) o[r] = (__bf16)acc[t][r];
    *reinterpret_cast<bf16x8*>(
        ONT + ((size_t)b * ERP_DIM + d) * N_TOK + n0 + mB) = o;
  }
}

// ---------------------------------------------------------------------------
extern "C" void kernel_launch(void* const* d_in, const int* in_sizes, int n_in,
                              void* d_out, int out_size, void* d_ws, size_t ws_size,
                              hipStream_t stream) {
  (void)in_sizes; (void)n_in; (void)out_size; (void)ws_size;
  const float* q      = (const float*)d_in[0];   // [4,4096,512]
  const float* kv     = (const float*)d_in[1];   // [4,4096,512]
  const float* q_w    = (const float*)d_in[2];   // [2,512,512]
  const float* kv_w   = (const float*)d_in[3];   // [2,1024,512]
  const float* proj_w = (const float*)d_in[4];   // [512,512]
  const float* proj_b = (const float*)d_in[5];   // [512]
  float* out = (float*)d_out;

  const size_t TOK = (size_t)BATCH * N_TOK * ERP_DIM;  // 8 Mi elems
  __bf16* qb  = (__bf16*)d_ws;         // q converted to bf16
  __bf16* kvb = qb  + TOK;             // kv converted
  __bf16* wq  = kvb + TOK;             // q_w  (2*512*512)
  __bf16* wkv = wq  + 2 * ERP_DIM * ERP_DIM;          // kv_w (2*1024*512)
  __bf16* wp  = wkv + 2 * 2 * ERP_DIM * ERP_DIM;      // proj_w
  __bf16* qp  = wp  + ERP_DIM * ERP_DIM;              // projected q
  __bf16* qn  = qp  + TOK;             // attention out (transposed) = next q
  __bf16* Kb  = qn  + TOK;             // K
  __bf16* VT  = Kb  + TOK;             // V transposed [b, d, m]

  auto cvt = [&](const float* s, __bf16* d, size_t n) {
    f32_to_bf16_kernel<<<dim3((unsigned)((n / 8 + 255) / 256)), dim3(256), 0,
                         stream>>>(s, d, (int)n);
  };
  cvt(q, qb, TOK);
  cvt(kv, kvb, TOK);
  cvt(q_w, wq, 2 * ERP_DIM * ERP_DIM);
  cvt(kv_w, wkv, 2 * 2 * ERP_DIM * ERP_DIM);
  cvt(proj_w, wp, ERP_DIM * ERP_DIM);

  const dim3 blk(128);
  const dim3 gProj(ERP_DIM / 64, (BATCH * N_TOK) / 64);       // (8, 256)
  const dim3 gKV((2 * ERP_DIM) / 64, (BATCH * N_TOK) / 64);   // (16, 256)
  const dim3 gAttn(BATCH * (N_TOK / 16));                     // 1024 blocks
  const size_t ldsBytes = (size_t)16 * N_TOK * sizeof(float); // 256 KB

  const __bf16* qcur = qb;
  for (int it = 0; it < 2; ++it) {
    gemm_bf16_wmma_kernel<true><<<gProj, blk, 0, stream>>>(
        qcur, wq + (size_t)it * ERP_DIM * ERP_DIM, nullptr, qp,
        ERP_DIM, ERP_DIM);
    gemm_kv_wmma_kernel<<<gKV, blk, 0, stream>>>(
        kvb, wkv + (size_t)it * 2 * ERP_DIM * ERP_DIM, Kb, VT);
    attn_lds_wmma_kernel<<<gAttn, dim3(256), ldsBytes, stream>>>(qp, Kb, VT, qn);
    qcur = qn;
  }
  gemm_bf16_wmma_kernel<false><<<gProj, blk, 0, stream>>>(
      qcur, wp, proj_b, out, ERP_DIM, ERP_DIM);
}
